// CrossAttention_27393301414116
// MI455X (gfx1250) — matrix operand order repaired and checked by reading, not compiled, not run on hardware
//
#include <hip/hip_runtime.h>
#include <hip/hip_bf16.h>

// ---------------------------------------------------------------------------
// CrossAttention on MI455X (gfx1250, wave32, WMMA bf16 + TDM LDS staging)
//   q = q_w @ q_feat        -> Qt (B, HW, C)  bf16   (transposed)
//   k = kv_w[0:C] @ kv_feat -> Kt (B, HW, C)  bf16   (transposed)
//   v = kv_w[C:2C]@ kv_feat -> V  (B, C, HW)  bf16   (natural)
//   fused flash attention (TDM-staged K/V tiles in LDS) -> Obf (B,C,HW) bf16
//   out = out_w @ Obf + out_b -> f32 (B, C, HW)
// ---------------------------------------------------------------------------

#define C_DIM 256
#define HW_DIM 4096
#define B_DIM 8

typedef __attribute__((ext_vector_type(16))) __bf16 v16bf;
typedef __attribute__((ext_vector_type(2)))  __bf16 v2bf;
typedef __attribute__((ext_vector_type(8)))  float  v8f;
typedef __attribute__((ext_vector_type(4)))  unsigned v4u;
typedef __attribute__((ext_vector_type(8)))  int      v8i;
typedef __attribute__((ext_vector_type(4)))  int      v4i;

union Frag {
    v16bf v;
    uint4 q[2];
    unsigned u[8];
    unsigned short s[16];
};
static_assert(sizeof(Frag) == 32, "frag must be 32B");

// native f32 -> bf16 (RNE): backend emits v_cvt_*bf16 ops
static __device__ __forceinline__ unsigned short f2bf(float f) {
    return __builtin_bit_cast(unsigned short, (__bf16)f);
}

static __device__ __forceinline__ unsigned pack2(float lo, float hi) {
    v2bf r;
    r.x = (__bf16)lo;
    r.y = (__bf16)hi;
    return __builtin_bit_cast(unsigned, r);
}

static __device__ __forceinline__ v8f wmma_bf16(v16bf a, v16bf b, v8f c) {
    // D(16x16,f32) = A(16x32,bf16) * B(32x16,bf16) + C
    return __builtin_amdgcn_wmma_f32_16x16x32_bf16(
        false, a, false, b, (short)0, c, false, false);
}

// ---------------------------------------------------------------------------
// Tensor Data Mover: 2D tile (bf16) global -> LDS.
// D# per CDNA5 ISA:
//   group0: [1:0]=count=1, [63:32]=lds_addr, [120:64]=global_addr, [127:126]=type=2
//   group1: [17:16]=data_size(1=2B), [79:48]=tensor_dim0, [111:80]=tensor_dim1,
//           [127:112]=tile_dim0, [143:128]=tile_dim1, [207:160]=tensor_dim0_stride
// This toolchain exposes the 6-arg builtin: (v4u, v8i, v4i, v4i, v8i, cpol).
// ---------------------------------------------------------------------------
static __device__ __forceinline__ void tdm_load_2d_bf16(
    const unsigned short* gptr, unsigned lds_addr,
    unsigned tensor_d0, unsigned tensor_d1,
    unsigned tile_d0, unsigned tile_d1, unsigned stride0)
{
    unsigned long long ga = (unsigned long long)(uintptr_t)gptr;
    v4u g0;
    g0[0] = 1u;                                              // count=1 (valid user D#)
    g0[1] = lds_addr;                                        // LDS byte address
    g0[2] = (unsigned)(ga & 0xFFFFFFFFu);                    // global_addr[31:0]
    g0[3] = (unsigned)((ga >> 32) & 0x01FFFFFFu) | (2u << 30); // addr[56:32] | type=2
    v8i g1;
    g1[0] = (int)(1u << 16);                                 // data_size=1 (2 bytes)
    g1[1] = (int)((tensor_d0 & 0xFFFFu) << 16);              // tensor_dim0[15:0]
    g1[2] = (int)(((tensor_d0 >> 16) & 0xFFFFu) | ((tensor_d1 & 0xFFFFu) << 16));
    g1[3] = (int)(((tensor_d1 >> 16) & 0xFFFFu) | ((tile_d0 & 0xFFFFu) << 16));
    g1[4] = (int)(tile_d1 & 0xFFFFu);                        // tile_dim1, tile_dim2=0
    g1[5] = (int)stride0;                                    // tensor_dim0_stride[31:0]
    g1[6] = 0;
    g1[7] = 0;
    v4i z4 = {0, 0, 0, 0};
    v8i z8 = {0, 0, 0, 0, 0, 0, 0, 0};
    __builtin_amdgcn_tensor_load_to_lds(g0, g1, z4, z4, z8, 0);
}

// ---------------------------------------------------------------------------
// Projection GEMM: Y[o,i] = sum_c W[o,c] * X[c,i] + bias[o]
// transposeOut = 1: store bf16 Y as (i, o); 0: natural (o, i)
// grid: (HW/128, C/16, B), block 256 (8 waves, one 16x16 tile each)
// ---------------------------------------------------------------------------
__global__ void proj_gemm(const float* __restrict__ X,
                          const float* __restrict__ W,
                          const float* __restrict__ bias,
                          unsigned short* __restrict__ Y,
                          int transposeOut) {
    const int lane    = threadIdx.x & 31;
    const int lane_lo = lane & 15;
    const int hi      = lane >> 4;
    const int wave    = threadIdx.x >> 5;

    const int o0 = blockIdx.y * 16;
    const int n0 = blockIdx.x * 128 + wave * 16;
    const int b  = blockIdx.z;

    const float* Xb = X + (size_t)b * C_DIM * HW_DIM;
    unsigned short* Yb = Y + (size_t)b * (size_t)C_DIM * HW_DIM;

    v8f acc = {};

    for (int c0 = 0; c0 < C_DIM; c0 += 32) {
        // A fragment: W rows o over K=c; elements[0..15]=c0+16h..
        Frag a;
        const float* wr = W + (size_t)(o0 + lane_lo) * C_DIM + c0 + 16 * hi;
#pragma unroll
        for (int k = 0; k < 4; ++k) a.u[k]     = pack2(wr[2 * k],      wr[2 * k + 1]);
#pragma unroll
        for (int k = 0; k < 4; ++k) a.u[4 + k] = pack2(wr[8 + 2 * k], wr[8 + 2 * k + 1]);

        // B fragment: X column i=n0+lane_lo, K=c strided by HW
        Frag bf;
        const float* xc = Xb + (size_t)(c0 + 16 * hi) * HW_DIM + (n0 + lane_lo);
#pragma unroll
        for (int k = 0; k < 8; ++k)
            bf.u[k] = pack2(xc[(size_t)(2 * k) * HW_DIM], xc[(size_t)(2 * k + 1) * HW_DIM]);

        acc = wmma_bf16(a.v, bf.v, acc);
    }

    if (transposeOut) {
        unsigned pk[4];
#pragma unroll
        for (int k = 0; k < 4; ++k) {
            float lo = acc[2 * k]     + bias[o0 + 2 * k     + 8 * hi];
            float hv = acc[2 * k + 1] + bias[o0 + 2 * k + 1 + 8 * hi];
            pk[k] = pack2(lo, hv);
        }
        uint4 st = make_uint4(pk[0], pk[1], pk[2], pk[3]);
        *(uint4*)(Yb + (size_t)(n0 + lane_lo) * C_DIM + o0 + 8 * hi) = st;
    } else {
#pragma unroll
        for (int v = 0; v < 8; ++v) {
            int o = o0 + v + 8 * hi;
            Yb[(size_t)o * HW_DIM + n0 + lane_lo] = f2bf(acc[v] + bias[o]);
        }
    }
}

// ---------------------------------------------------------------------------
// Fused flash attention with TDM double-buffered K/V staging in LDS.
//   Per block: 8 waves, same batch, 8 consecutive 16-query tiles.
//   Per 32-j step: wave0 TDM-loads K tile (32x256) and V tile (256x32) into
//   LDS (prefetching next step), all waves read WMMA fragments from LDS.
// grid: (HW/128, B), block 256
// ---------------------------------------------------------------------------
__global__ __launch_bounds__(256) void flash_attn(
        const unsigned short* __restrict__ Qt,
        const unsigned short* __restrict__ Kt,
        const unsigned short* __restrict__ V,
        unsigned short* __restrict__ Obf) {
    __shared__ unsigned short Kbuf[2][32 * C_DIM];   // 2 x 16KB
    __shared__ unsigned short Vbuf[2][C_DIM * 32];   // 2 x 16KB

    const int lane    = threadIdx.x & 31;
    const int lane_lo = lane & 15;
    const int hi      = lane >> 4;
    const int wave    = threadIdx.x >> 5;
    const int b       = blockIdx.y;
    const int i0      = (blockIdx.x * 8 + wave) * 16;

    const unsigned short* Qb = Qt + (size_t)b * HW_DIM * C_DIM;
    const unsigned short* Kb = Kt + (size_t)b * HW_DIM * C_DIM;
    const unsigned short* Vb = V  + (size_t)b * (size_t)C_DIM * HW_DIM;
    unsigned short*       Ob = Obf + (size_t)b * (size_t)C_DIM * HW_DIM;

    const unsigned klds[2] = { (unsigned)(uintptr_t)&Kbuf[0][0],
                               (unsigned)(uintptr_t)&Kbuf[1][0] };
    const unsigned vlds[2] = { (unsigned)(uintptr_t)&Vbuf[0][0],
                               (unsigned)(uintptr_t)&Vbuf[1][0] };

    // Preload loop-invariant Q B-fragments: column i = i0+lane_lo over K=c
    Frag qb[8];
#pragma unroll
    for (int cc = 0; cc < 8; ++cc) {
        const unsigned short* qr = Qb + (size_t)(i0 + lane_lo) * C_DIM + cc * 32 + 8 * hi;
        qb[cc].q[0] = *(const uint4*)qr;
        qb[cc].q[1] = *(const uint4*)(qr + 16);
    }

    v8f Oacc[16];
#pragma unroll
    for (int t = 0; t < 16; ++t) Oacc[t] = (v8f){};

    float m_run = -1e30f;
    float l_run = 0.0f;
    const float SCALE = 0.0625f;   // 1/sqrt(256)
    const int nsteps = HW_DIM / 32;  // 128

    // prologue: stage step 0
    if (wave == 0) {
        // K tile: tensor (4096 x 256), tile 32 rows x 256, contiguous rows
        tdm_load_2d_bf16(Kb, klds[0], C_DIM, HW_DIM, C_DIM, 32, C_DIM);
        // V tile: tensor (256 x 4096), tile 256 rows x 32 cols
        tdm_load_2d_bf16(Vb, vlds[0], HW_DIM, C_DIM, 32, C_DIM, HW_DIM);
    }

    for (int s = 0; s < nsteps; ++s) {
        const int cur = s & 1;
        // prefetch next step into the other buffer (it was last read in step s-1,
        // and the end-of-step barrier of s-1 has already been passed)
        if (wave == 0) {
            if (s + 1 < nsteps) {
                const int j1 = (s + 1) * 32;
                tdm_load_2d_bf16(Kb + (size_t)j1 * C_DIM, klds[cur ^ 1],
                                 C_DIM, HW_DIM, C_DIM, 32, C_DIM);
                tdm_load_2d_bf16(Vb + j1, vlds[cur ^ 1],
                                 HW_DIM, C_DIM, 32, C_DIM, HW_DIM);
                __builtin_amdgcn_s_wait_tensorcnt(2);  // pair s done, pair s+1 in flight
            } else {
                __builtin_amdgcn_s_wait_tensorcnt(0);
            }
        }
        __syncthreads();   // publish buffer `cur` to all waves

        const int j0 = s * 32;
        (void)j0;

        // ---- two S^T tiles (rows j, cols i) from LDS K + register Q ----
        v8f s0 = {}, s1 = {};
#pragma unroll
        for (int cc = 0; cc < 8; ++cc) {
            Frag a0, a1;
            const unsigned short* kr0 = &Kbuf[cur][lane_lo * C_DIM + cc * 32 + 8 * hi];
            a0.q[0] = *(const uint4*)kr0;
            a0.q[1] = *(const uint4*)(kr0 + 16);
            const unsigned short* kr1 = kr0 + 16 * C_DIM;
            a1.q[0] = *(const uint4*)kr1;
            a1.q[1] = *(const uint4*)(kr1 + 16);
            s0 = wmma_bf16(a0.v, qb[cc].v, s0);
            s1 = wmma_bf16(a1.v, qb[cc].v, s1);
        }

        // ---- online softmax over the 32 j's (column i lives in one lane pair) ----
        float p0[8], p1[8];
        float tm = -1e30f;
#pragma unroll
        for (int v = 0; v < 8; ++v) {
            p0[v] = s0[v] * SCALE;
            p1[v] = s1[v] * SCALE;
            tm = fmaxf(tm, fmaxf(p0[v], p1[v]));
        }
        tm = fmaxf(tm, __shfl_xor(tm, 16, 32));
        float m_new = fmaxf(m_run, tm);
        float corr  = __expf(m_run - m_new);
        float ls = 0.0f;
#pragma unroll
        for (int v = 0; v < 8; ++v) {
            p0[v] = __expf(p0[v] - m_new);
            p1[v] = __expf(p1[v] - m_new);
            ls += p0[v] + p1[v];
        }
        ls += __shfl_xor(ls, 16, 32);
        l_run = l_run * corr + ls;
        m_run = m_new;

        // rescale accumulator: per-lane scalar (lane owns one query column)
#pragma unroll
        for (int t = 0; t < 16; ++t)
#pragma unroll
            for (int v = 0; v < 8; ++v) Oacc[t][v] *= corr;

        // ---- build P^T B-fragment (32x16, K=j) from the two C/D-layout tiles ----
        unsigned pk0[4], pk1[4], ox0[4], ox1[4];
#pragma unroll
        for (int k = 0; k < 4; ++k) {
            pk0[k] = pack2(p0[2 * k], p0[2 * k + 1]);
            pk1[k] = pack2(p1[2 * k], p1[2 * k + 1]);
        }
#pragma unroll
        for (int k = 0; k < 4; ++k) {
            ox0[k] = __shfl_xor(pk0[k], 16, 32);
            ox1[k] = __shfl_xor(pk1[k], 16, 32);
        }
        Frag pb;
        bool upper = (hi != 0);
#pragma unroll
        for (int k = 0; k < 4; ++k) {
            pb.u[k]     = upper ? ox1[k] : pk0[k];   // tile rows 0..7
            pb.u[4 + k] = upper ? pk1[k] : ox0[k];   // tile rows 8..15
        }

        // ---- O += V(:, j0..j0+31) * P^T : 16 c-tiles, A from LDS ----
#pragma unroll
        for (int t = 0; t < 16; ++t) {
            Frag a;
            const unsigned short* vr = &Vbuf[cur][(t * 16 + lane_lo) * 32 + 16 * hi];
            a.q[0] = *(const uint4*)vr;
            a.q[1] = *(const uint4*)(vr + 8);
            Oacc[t] = wmma_bf16(a.v, pb.v, Oacc[t]);
        }

        __syncthreads();   // all waves done reading `cur` before it is re-staged
    }

    // ---- epilogue: divide by l, store bf16 natural (c, i) ----
    float invl = 1.0f / l_run;
#pragma unroll
    for (int t = 0; t < 16; ++t)
#pragma unroll
        for (int v = 0; v < 8; ++v) {
            int c = t * 16 + v + 8 * hi;
            Ob[(size_t)c * HW_DIM + i0 + lane_lo] = f2bf(Oacc[t][v] * invl);
        }
}

// ---------------------------------------------------------------------------
// Final conv: out[o,i] = sum_c out_w[o,c] * Obf[c,i] + out_b[o]   (f32 out)
// grid: (HW/128, C/16, B), block 256
// ---------------------------------------------------------------------------
__global__ void final_gemm(const unsigned short* __restrict__ Obf,
                           const float* __restrict__ W,
                           const float* __restrict__ bias,
                           float* __restrict__ OutF) {
    const int lane    = threadIdx.x & 31;
    const int lane_lo = lane & 15;
    const int hi      = lane >> 4;
    const int wave    = threadIdx.x >> 5;

    const int o0 = blockIdx.y * 16;
    const int n0 = blockIdx.x * 128 + wave * 16;
    const int b  = blockIdx.z;

    const unsigned short* Ob = Obf + (size_t)b * (size_t)C_DIM * HW_DIM;
    float* Yb = OutF + (size_t)b * (size_t)C_DIM * HW_DIM;

    v8f acc = {};

    for (int c0 = 0; c0 < C_DIM; c0 += 32) {
        Frag a;
        const float* wr = W + (size_t)(o0 + lane_lo) * C_DIM + c0 + 16 * hi;
#pragma unroll
        for (int k = 0; k < 4; ++k) a.u[k]     = pack2(wr[2 * k],      wr[2 * k + 1]);
#pragma unroll
        for (int k = 0; k < 4; ++k) a.u[4 + k] = pack2(wr[8 + 2 * k], wr[8 + 2 * k + 1]);

        Frag bf;
        const unsigned short* oc = Ob + (size_t)(c0 + 16 * hi) * HW_DIM + (n0 + lane_lo);
#pragma unroll
        for (int k = 0; k < 8; ++k)
            bf.u[k] = (unsigned)oc[(size_t)(2 * k) * HW_DIM] |
                      ((unsigned)oc[(size_t)(2 * k + 1) * HW_DIM] << 16);

        acc = wmma_bf16(a.v, bf.v, acc);
    }

#pragma unroll
    for (int v = 0; v < 8; ++v) {
        int o = o0 + v + 8 * hi;
        Yb[(size_t)o * HW_DIM + n0 + lane_lo] = acc[v] + bias[o];
    }
}

// ---------------------------------------------------------------------------
extern "C" void kernel_launch(void* const* d_in, const int* in_sizes, int n_in,
                              void* d_out, int out_size, void* d_ws, size_t ws_size,
                              hipStream_t stream) {
    (void)in_sizes; (void)n_in; (void)out_size; (void)ws_size;

    const float* q_feat  = (const float*)d_in[0];
    const float* kv_feat = (const float*)d_in[1];
    const float* q_w     = (const float*)d_in[2];
    const float* q_b     = (const float*)d_in[3];
    const float* kv_w    = (const float*)d_in[4];
    const float* kv_b    = (const float*)d_in[5];
    const float* out_w   = (const float*)d_in[6];
    const float* out_b   = (const float*)d_in[7];
    float* out = (float*)d_out;

    const size_t N = (size_t)B_DIM * HW_DIM * C_DIM;   // 8M elems, 16MB bf16 each
    unsigned short* Qt  = (unsigned short*)d_ws;
    unsigned short* Kt  = Qt + N;
    unsigned short* Vn  = Kt + N;
    unsigned short* Obf = Vn + N;

    dim3 blk(256);
    dim3 gp(HW_DIM / 128, C_DIM / 16, B_DIM);

    // projections
    proj_gemm<<<gp, blk, 0, stream>>>(q_feat,  q_w,              q_b,        Qt, 1);
    proj_gemm<<<gp, blk, 0, stream>>>(kv_feat, kv_w,             kv_b,       Kt, 1);
    proj_gemm<<<gp, blk, 0, stream>>>(kv_feat, kv_w + 256 * 256, kv_b + 256, Vn, 0);

    // fused attention with TDM staging
    flash_attn<<<dim3(HW_DIM / 128, B_DIM), blk, 0, stream>>>(Qt, Kt, Vn, Obf);

    // output projection
    final_gemm<<<gp, blk, 0, stream>>>(Obf, out_w, out_b, out);
}
